// SeqEncoder_40819369181477
// MI455X (gfx1250) — matrix operand order repaired and checked by reading, not compiled, not run on hardware
//
#include <hip/hip_runtime.h>
#include <hip/hip_bf16.h>

// ---------------------------------------------------------------------------
// GRU SeqEncoder for MI455X (gfx1250, wave32, WMMA bf16 16x16x32, f32 accum)
// S=512, B=64, E=512, H=1024, 2 layers.
// ---------------------------------------------------------------------------

#define GS   512      // sequence length
#define GB   64       // batch
#define GE   512      // embed dim
#define GH   1024     // hidden dim
#define G3H  3072     // 3*H

#define WROW 1032     // LDS weight row stride in bf16 elems (2064B = 516 dwords)
#define WLDS_BYTES (48 * WROW * 2)               // 3 gates x 16 rows
#define SH_BYTES   (3 * GB * 16 * 4)             // gate-exchange tiles
#define REC_SMEM   (WLDS_BYTES + SH_BYTES)       // 111360 B (< 320KB WGP LDS)

#define TROW 40       // GEMM LDS tile row stride in bf16 (80B = 20 dwords)

typedef __attribute__((ext_vector_type(16))) __bf16 v16bf;
typedef __attribute__((ext_vector_type(8)))  float  v8f;

union ABfrag { v16bf v; uint4 u[2]; };

__device__ __forceinline__ float sigm(float x) { return 1.0f / (1.0f + __expf(-x)); }

// ---------------- grid-wide sense-reversing barrier (persistent kernels) ----
__device__ __forceinline__ void grid_barrier(int* cnt, int* gen, int nwg) {
  __syncthreads();
  if (threadIdx.x == 0) {
    __threadfence();
    int g = __hip_atomic_load(gen, __ATOMIC_ACQUIRE, __HIP_MEMORY_SCOPE_AGENT);
    if (__hip_atomic_fetch_add(cnt, 1, __ATOMIC_ACQ_REL, __HIP_MEMORY_SCOPE_AGENT) == nwg - 1) {
      __hip_atomic_store(cnt, 0, __ATOMIC_RELAXED, __HIP_MEMORY_SCOPE_AGENT);
      __hip_atomic_fetch_add(gen, 1, __ATOMIC_ACQ_REL, __HIP_MEMORY_SCOPE_AGENT);
    } else {
      while (__hip_atomic_load(gen, __ATOMIC_ACQUIRE, __HIP_MEMORY_SCOPE_AGENT) == g) {
        __builtin_amdgcn_s_sleep(1);
      }
    }
    __threadfence();
  }
  __syncthreads();
}

// ---------------- f32 -> bf16 convert (weights) -----------------------------
__global__ __launch_bounds__(256)
void cvt_bf16_kernel(const float* __restrict__ in, __hip_bfloat16* __restrict__ out, int n) {
  for (int i = blockIdx.x * 256 + threadIdx.x; i < n; i += gridDim.x * 256)
    out[i] = __float2bfloat16(in[i]);
}

// ---------------- embedding gather + bf16 -----------------------------------
__global__ __launch_bounds__(256)
void embed_kernel(const int* __restrict__ x, const float* __restrict__ emb,
                  __hip_bfloat16* __restrict__ out) {
  const int row = blockIdx.x;              // row = s*B + b   (32768 rows)
  const int idx = x[row];
  const float* src = emb + (size_t)idx * GE;
  __hip_bfloat16* dst = out + (size_t)row * GE;
  for (int e = threadIdx.x; e < GE; e += 256) dst[e] = __float2bfloat16(src[e]);
}

// ---------------- x_proj GEMM: Out(M,N) = X(M,K) * W(N,K)^T + bias ----------
// bf16 inputs, f32 accumulate. Two-stage pipeline:
//   iter kt: ds_store regs(kt, loaded at kt-1) -> LDS[buf]; sync;
//            global_load regs(kt+1);   [full iteration of L2 latency slack]
//            ds_load fragments + 4 WMMAs from LDS[buf].
// One __syncthreads per k-step; WMMA dscnt waits only cover local LDS traffic.
__global__ __launch_bounds__(256)
void gemm_xproj_kernel(const __hip_bfloat16* __restrict__ X,
                       const __hip_bfloat16* __restrict__ W,
                       const float* __restrict__ bias,
                       float* __restrict__ Out,
                       int M, int N, int K) {
  __shared__ __align__(16) __hip_bfloat16 Ash[2][64 * TROW];    // 10 KB
  __shared__ __align__(16) __hip_bfloat16 Bsh[2][128 * TROW];   // 20 KB

  const int tid  = threadIdx.x;
  const int lane = tid & 31;
  const int wid  = tid >> 5;
  const int mblk = blockIdx.y * 64;
  const int nblk = blockIdx.x * 128;
  const int mrow0 = mblk + (wid & 3) * 16;
  const int ncl   = (wid >> 2) * 64;          // ngroup column offset in tile

  v8f acc[4] = {v8f{}, v8f{}, v8f{}, v8f{}};

  const int n15 = lane & 15;
  const int kbA = (lane & 16) ? 16 : 0;       // contiguous A frag K chunk (k = 16h + s)
  const int k0B = (lane & 16) ? 16 : 0;       // contiguous B frag K chunk (same k order as A)

  // staging geometry: row-major [row][k] with TROW padding
  const int srowA = tid >> 2, schA = (tid & 3) * 8;               // 256 uint4
  const __hip_bfloat16* gA = X + (size_t)(mblk + srowA) * K + schA;
  const int srB0 = tid >> 2, srB1 = (tid + 256) >> 2;             // 512 uint4
  const int scB0 = (tid & 3) * 8, scB1 = ((tid + 256) & 3) * 8;
  const __hip_bfloat16* gB0 = W + (size_t)(nblk + srB0) * K + scB0;
  const __hip_bfloat16* gB1 = W + (size_t)(nblk + srB1) * K + scB1;

  // prologue: tile k=0 into staging registers
  uint4 ra  = *(const uint4*)(gA);
  uint4 rb0 = *(const uint4*)(gB0);
  uint4 rb1 = *(const uint4*)(gB1);

  const int KT = K >> 5;
  int buf = 0;
  for (int kt = 0; kt < KT; ++kt) {
    // stage tile kt from registers into LDS[buf]
    *(uint4*)(Ash[buf] + srowA * TROW + schA) = ra;
    *(uint4*)(Bsh[buf] + srB0 * TROW + scB0)  = rb0;
    *(uint4*)(Bsh[buf] + srB1 * TROW + scB1)  = rb1;
    __syncthreads();

    // issue global loads for tile kt+1 (clamped; consumed next iteration)
    const int kn = (kt + 1 < KT) ? ((kt + 1) << 5) : (kt << 5);
    ra  = *(const uint4*)(gA + kn);
    rb0 = *(const uint4*)(gB0 + kn);
    rb1 = *(const uint4*)(gB1 + kn);

    // fragment reads first (a + all 4 b), then the 4 WMMAs
    const __hip_bfloat16* aptr = Ash[buf] + ((wid & 3) * 16 + n15) * TROW + kbA;
    ABfrag a, b[4];
    a.u[0] = *(const uint4*)(aptr);
    a.u[1] = *(const uint4*)(aptr + 8);
#pragma unroll
    for (int nt = 0; nt < 4; ++nt) {
      const __hip_bfloat16* bptr = Bsh[buf] + (ncl + nt * 16 + n15) * TROW + k0B;
      b[nt].u[0] = ((const uint4*)bptr)[0];
      b[nt].u[1] = ((const uint4*)bptr)[1];
    }
#pragma unroll
    for (int nt = 0; nt < 4; ++nt)
      acc[nt] = __builtin_amdgcn_wmma_f32_16x16x32_bf16(
          false, a.v, false, b[nt].v, (short)0, acc[nt], false, false);

    buf ^= 1;   // buf^1 was last read at kt-1; kt's barrier separates reuse
  }

  // C layout: lane n = lane&15; element r -> M = r + 8*(lane>=16)
  const int mg = mrow0 + ((lane & 16) ? 8 : 0);
#pragma unroll
  for (int nt = 0; nt < 4; ++nt) {
    const int col = nblk + ncl + nt * 16 + n15;
    const float bv = bias[col];
#pragma unroll
    for (int r = 0; r < 8; ++r)
      Out[(size_t)(mg + r) * N + col] = acc[nt][r] + bv;
  }
}

// ---------------- persistent GRU recurrence ---------------------------------
// 64 WGs x 128 threads. WG owns 16 hidden columns. w_hh slice (3x16x1024 bf16
// = 96KB) is staged ONCE in LDS (CDNA5: 320KB/WGP) and re-read by ds_load each
// of the 512 steps; hidden-state A tiles stream from L2, double-buffered in
// registers across the k-loop. One grid barrier per timestep.
__global__ __launch_bounds__(128)
void gru_rec_kernel(const float* __restrict__ Xp,             // (S,B,3H) f32
                    const __hip_bfloat16* __restrict__ Whh,   // (3H,H) bf16 row-major
                    const float* __restrict__ bhh,            // (3H)
                    float* __restrict__ hstate,               // (B,H) f32 -> final hidden
                    __hip_bfloat16* __restrict__ hbf0,        // (B,H) bf16 ping
                    __hip_bfloat16* __restrict__ hbf1,        // (B,H) bf16 pong
                    float* __restrict__ Yf,                   // (S,B,H) f32 or null
                    __hip_bfloat16* __restrict__ Ybf,         // (S,B,H) bf16 or null
                    int* __restrict__ bar, int S) {
  extern __shared__ char smem[];
  __hip_bfloat16* wlds = (__hip_bfloat16*)smem;               // [48][WROW]
  float (*sh)[GB][16] = (float (*)[GB][16])(smem + WLDS_BYTES);

  const int tid  = threadIdx.x;
  const int lane = tid & 31;
  const int wid  = tid >> 5;            // 0..3 = batch tile
  const int jb   = blockIdx.x * 16;     // this WG's hidden-column block
  int* cnt = bar;
  int* gen = bar + 1;

  // ---- stage w_hh slice into LDS: 48 rows (gate-major), padded stride ----
  for (int idx = tid; idx < 48 * (GH / 8); idx += 128) {   // 6144 uint4 chunks
    const int r = idx >> 7, ch = idx & 127;                // row, 16B chunk
    const int g = r >> 4, n = r & 15;
    const uint4 v = ((const uint4*)(Whh + (size_t)(g * GH + jb + n) * GH))[ch];
    *(uint4*)(wlds + r * WROW + ch * 8) = v;
  }

  // zero-init owned slice of the state (also clears harness poison)
  for (int idx = tid; idx < GB * 16; idx += 128) {
    const int m2 = idx >> 4, c = idx & 15;
    hstate[(size_t)m2 * GH + jb + c] = 0.0f;
    hbf0[(size_t)m2 * GH + jb + c]   = __float2bfloat16(0.0f);
  }
  grid_barrier(cnt, gen, gridDim.x);    // also covers LDS staging (syncthreads)

  const int mbase = wid * 16;
  const int n15   = lane & 15;
  const int m     = mbase + n15;
  const int kbA   = (lane & 16) ? 16 : 0;
  const int k0B   = (lane & 16) ? 16 : 0;
  const __hip_bfloat16* w0 = wlds + (0 * 16 + n15) * WROW + k0B;
  const __hip_bfloat16* w1 = wlds + (1 * 16 + n15) * WROW + k0B;
  const __hip_bfloat16* w2 = wlds + (2 * 16 + n15) * WROW + k0B;

  for (int t = 0; t < S; ++t) {
    const __hip_bfloat16* hin  = (t & 1) ? hbf1 : hbf0;
    __hip_bfloat16*       hout = (t & 1) ? hbf0 : hbf1;

    v8f acc0 = {}, acc1 = {}, acc2 = {};
    const __hip_bfloat16* arow = hin + (size_t)m * GH;

    ABfrag a;
    a.u[0] = *(const uint4*)(arow + kbA);
    a.u[1] = *(const uint4*)(arow + kbA + 8);

    for (int k = 0; k < GH; k += 32) {
      const int kn = (k + 32 < GH) ? (k + 32) : k;
      ABfrag an;                                  // prefetch next A from L2
      an.u[0] = *(const uint4*)(arow + kn + kbA);
      an.u[1] = *(const uint4*)(arow + kn + kbA + 8);
      ABfrag b0, b1, b2;                          // B from LDS (ds_load_b128)
      b0.u[0] = ((const uint4*)(w0 + k))[0];
      b0.u[1] = ((const uint4*)(w0 + k))[1];
      b1.u[0] = ((const uint4*)(w1 + k))[0];
      b1.u[1] = ((const uint4*)(w1 + k))[1];
      b2.u[0] = ((const uint4*)(w2 + k))[0];
      b2.u[1] = ((const uint4*)(w2 + k))[1];
      acc0 = __builtin_amdgcn_wmma_f32_16x16x32_bf16(false, a.v, false, b0.v,
                                                     (short)0, acc0, false, false);
      acc1 = __builtin_amdgcn_wmma_f32_16x16x32_bf16(false, a.v, false, b1.v,
                                                     (short)0, acc1, false, false);
      acc2 = __builtin_amdgcn_wmma_f32_16x16x32_bf16(false, a.v, false, b2.v,
                                                     (short)0, acc2, false, false);
      a = an;
    }

    // scatter gate tiles to LDS (C layout: row = r + 8*(lane>=16), col = lane&15)
#pragma unroll
    for (int r = 0; r < 8; ++r) {
      const int mm = mbase + r + ((lane & 16) ? 8 : 0);
      sh[0][mm][n15] = acc0[r];
      sh[1][mm][n15] = acc1[r];
      sh[2][mm][n15] = acc2[r];
    }
    __syncthreads();

    // elementwise GRU update for owned (m, j) cells
    for (int idx = tid; idx < GB * 16; idx += 128) {
      const int m2 = idx >> 4, c = idx & 15, j = jb + c;
      const float ghr = sh[0][m2][c] + bhh[j];
      const float ghz = sh[1][m2][c] + bhh[GH + j];
      const float ghn = sh[2][m2][c] + bhh[2 * GH + j];
      const size_t xb = ((size_t)t * GB + m2) * G3H + j;
      const float r_ = sigm(Xp[xb] + ghr);
      const float z_ = sigm(Xp[xb + GH] + ghz);
      const float n_ = tanhf(Xp[xb + 2 * GH] + r_ * ghn);
      const size_t hi = (size_t)m2 * GH + j;
      const float hn = (1.0f - z_) * n_ + z_ * hstate[hi];
      hstate[hi] = hn;
      hout[hi]   = __float2bfloat16(hn);
      const size_t yb = ((size_t)t * GB + m2) * GH + j;
      if (Yf)  Yf[yb]  = hn;
      if (Ybf) Ybf[yb] = __float2bfloat16(hn);
    }
    grid_barrier(cnt, gen, gridDim.x);   // writes of hout visible before t+1
  }
}

// ---------------------------------------------------------------------------
extern "C" void kernel_launch(void* const* d_in, const int* in_sizes, int n_in,
                              void* d_out, int out_size, void* d_ws, size_t ws_size,
                              hipStream_t stream) {
  (void)in_sizes; (void)n_in; (void)out_size; (void)ws_size;
  const int*   x     = (const int*)d_in[0];
  const float* emb   = (const float*)d_in[1];
  const float* w_ih0 = (const float*)d_in[2];
  const float* w_hh0 = (const float*)d_in[3];
  const float* b_ih0 = (const float*)d_in[4];
  const float* b_hh0 = (const float*)d_in[5];
  const float* w_ih1 = (const float*)d_in[6];
  const float* w_hh1 = (const float*)d_in[7];
  const float* b_ih1 = (const float*)d_in[8];
  const float* b_hh1 = (const float*)d_in[9];
  float* out = (float*)d_out;

  const int MROWS = GS * GB;          // 32768

  // workspace carve-up (~502 MB total)
  size_t off = 0;
  auto carve = [&](size_t bytes) { size_t o = off; off = (off + bytes + 255) & ~(size_t)255; return o; };
  char* ws = (char*)d_ws;
  float*           Xp    = (float*)(ws + carve((size_t)MROWS * G3H * 4)); // 402 MB, reused per layer
  __hip_bfloat16*  X0bf  = (__hip_bfloat16*)(ws + carve((size_t)MROWS * GE * 2));
  __hip_bfloat16*  Y0bf  = (__hip_bfloat16*)(ws + carve((size_t)MROWS * GH * 2));
  __hip_bfloat16*  Wih0b = (__hip_bfloat16*)(ws + carve((size_t)G3H * GE * 2));
  __hip_bfloat16*  Whh0b = (__hip_bfloat16*)(ws + carve((size_t)G3H * GH * 2));
  __hip_bfloat16*  Wih1b = (__hip_bfloat16*)(ws + carve((size_t)G3H * GH * 2));
  __hip_bfloat16*  Whh1b = (__hip_bfloat16*)(ws + carve((size_t)G3H * GH * 2));
  __hip_bfloat16*  hbf0  = (__hip_bfloat16*)(ws + carve((size_t)GB * GH * 2));
  __hip_bfloat16*  hbf1  = (__hip_bfloat16*)(ws + carve((size_t)GB * GH * 2));
  int*             bar   = (int*)(ws + carve(256));

  // final hidden slots live directly in d_out (recurrence state in place)
  float* hfinal0 = out + (size_t)GS * GB * GH;
  float* hfinal1 = hfinal0 + (size_t)GB * GH;
  float* Y1f32   = out;                      // layer-1 outputs (S,B,H)

  // 1) weight converts to bf16
  cvt_bf16_kernel<<<1024, 256, 0, stream>>>(w_ih0, Wih0b, G3H * GE);
  cvt_bf16_kernel<<<1024, 256, 0, stream>>>(w_hh0, Whh0b, G3H * GH);
  cvt_bf16_kernel<<<1024, 256, 0, stream>>>(w_ih1, Wih1b, G3H * GH);
  cvt_bf16_kernel<<<1024, 256, 0, stream>>>(w_hh1, Whh1b, G3H * GH);

  // 2) embedding gather -> bf16 rows
  embed_kernel<<<MROWS, 256, 0, stream>>>(x, emb, X0bf);

  // 3) layer-0 input projection: Xp = X0 @ w_ih0^T + b_ih0
  gemm_xproj_kernel<<<dim3(G3H / 128, MROWS / 64), 256, 0, stream>>>(
      X0bf, Wih0b, b_ih0, Xp, MROWS, G3H, GE);

  // 4) layer-0 recurrence (persistent, grid-synced); emits bf16 seq for layer 1
  hipMemsetAsync(bar, 0, 256, stream);
  gru_rec_kernel<<<GH / 16, 128, REC_SMEM, stream>>>(Xp, Whh0b, b_hh0, hfinal0,
                                                     hbf0, hbf1, nullptr, Y0bf, bar, GS);

  // 5) layer-1 input projection: Xp = Y0 @ w_ih1^T + b_ih1
  gemm_xproj_kernel<<<dim3(G3H / 128, MROWS / 64), 256, 0, stream>>>(
      Y0bf, Wih1b, b_ih1, Xp, MROWS, G3H, GH);

  // 6) layer-1 recurrence; f32 outputs straight to d_out
  hipMemsetAsync(bar, 0, 256, stream);
  gru_rec_kernel<<<GH / 16, 128, REC_SMEM, stream>>>(Xp, Whh1b, b_hh1, hfinal1,
                                                     hbf0, hbf1, Y1f32, nullptr, bar, GS);
}
